// GCN_convblk_53403623358890
// MI455X (gfx1250) — compile-verified, matrix operand order checked by reading, not run on hardware
//
#include <hip/hip_runtime.h>

#define NNODES 50000
#define NEDGES 400000
#define FDIM   256
#define HDIM   512
#define CDIM   3
#define NBLK   6

typedef __attribute__((ext_vector_type(16))) _Float16 v16h;
typedef __attribute__((ext_vector_type(8)))  _Float16 v8h;
typedef __attribute__((ext_vector_type(8)))  float    v8f;
typedef __attribute__((ext_vector_type(4)))  float    v4f;

// ---------------------------------------------------------------------------
// CSR construction (run once per launch; reused by all 14 layers)
// ---------------------------------------------------------------------------
__global__ void k_fill_zero_i32(int* p, int n) {
    int t = blockIdx.x * blockDim.x + threadIdx.x;
    if (t < n) p[t] = 0;
}

__global__ void k_count_deg(const int* __restrict__ dst, int* __restrict__ cnt, int e) {
    int t = blockIdx.x * blockDim.x + threadIdx.x;
    if (t < e) atomicAdd(&cnt[dst[t]], 1);
}

__global__ void k_dinv(const int* __restrict__ cnt, float* __restrict__ dinv, int n) {
    int t = blockIdx.x * blockDim.x + threadIdx.x;
    if (t < n) dinv[t] = rsqrtf((float)cnt[t] + 1.0f);   // +1 self loop
}

// single-block exclusive scan (Hillis-Steele over 1024-wide chunks)
__global__ __launch_bounds__(1024) void k_scan_excl(const int* __restrict__ cnt,
                                                    int* __restrict__ rowptr, int n) {
    __shared__ int tmp[1024];
    __shared__ int carry;
    if (threadIdx.x == 0) carry = 0;
    __syncthreads();
    for (int base = 0; base < n; base += 1024) {
        int idx = base + (int)threadIdx.x;
        int v = (idx < n) ? cnt[idx] : 0;
        tmp[threadIdx.x] = v;
        __syncthreads();
        #pragma unroll
        for (int off = 1; off < 1024; off <<= 1) {
            int t = (threadIdx.x >= (unsigned)off) ? tmp[threadIdx.x - off] : 0;
            __syncthreads();
            tmp[threadIdx.x] += t;
            __syncthreads();
        }
        if (idx < n) rowptr[idx] = carry + tmp[threadIdx.x] - v;
        __syncthreads();
        if (threadIdx.x == 1023) carry += tmp[1023];
        __syncthreads();
    }
    if (threadIdx.x == 0) rowptr[n] = carry;
}

__global__ void k_fill_csr(const int* __restrict__ src, const int* __restrict__ dst,
                           const int* __restrict__ rowptr, int* __restrict__ cur,
                           const float* __restrict__ dinv,
                           int* __restrict__ colb, float* __restrict__ ew, int e) {
    int t = blockIdx.x * blockDim.x + threadIdx.x;
    if (t >= e) return;
    int s = src[t], d = dst[t];
    int p = rowptr[d] + atomicAdd(&cur[d], 1);
    colb[p] = s;
    ew[p]   = dinv[s] * dinv[d];
}

// ---------------------------------------------------------------------------
// Weight prep: W [K][NC] f32 row-major -> Wt [NC][K] f16 (so the WMMA B
// operand per lane is two contiguous 16-byte runs)
// ---------------------------------------------------------------------------
__global__ void k_prep_w(const float* __restrict__ W, _Float16* __restrict__ Wt,
                         int K, int NC) {
    int t = blockIdx.x * blockDim.x + threadIdx.x;
    if (t >= K * NC) return;
    int k = t / NC, n = t - k * NC;
    Wt[(long)n * K + k] = (_Float16)W[t];
}

// ---------------------------------------------------------------------------
// WMMA GEMM:  C[M][NC] f32 = A[M][K] f32 (optional fused ReLU, cvt->f16)
//                          @ Wt[NC][K] f16 (pre-transposed)
// one wave per 16x64 output tile: a single A fragment is reused by 4
// back-to-back v_wmma_f32_16x16x32_f16 against 4 B fragments, cutting
// A-side L2 traffic 4x vs wave-per-16x16.
// A-operand layout (ISA 7.12.2, 16-bit A 16x32): lane&15 = row,
// lane>>4 selects K-halves {0..7,16..23} vs {8..15,24..31}.
// ---------------------------------------------------------------------------
template <bool RELU, int K>
__global__ __launch_bounds__(256) void k_gemm_wmma(const float* __restrict__ A,
                                                   const _Float16* __restrict__ Bt,
                                                   float* __restrict__ C,
                                                   int M, int NC) {
    const int wave = blockIdx.x * (blockDim.x >> 5) + (threadIdx.x >> 5);
    const int lane = threadIdx.x & 31;
    const int tiles_n = NC >> 6;              // 64-wide column tiles
    const int mt = wave / tiles_n;
    const int nt = wave - mt * tiles_n;
    if (mt >= (M >> 4)) return;

    const int rlo = lane & 15;
    const int hi  = lane >> 4;                // 0 or 1

    const float*    arow = A  + (long)(mt * 16 + rlo) * K + hi * 8;
    const _Float16* brow = Bt + (long)(nt * 64 + rlo) * K + hi * 8;

    v8f acc[4] = {v8f{}, v8f{}, v8f{}, v8f{}};

    #pragma unroll 4
    for (int k = 0; k < K; k += 32) {
        __builtin_prefetch(arow + k + 32, 0, 1);   // global_prefetch_b8
        v4f a0 = *(const v4f*)(arow + k);
        v4f a1 = *(const v4f*)(arow + k + 4);
        v4f a2 = *(const v4f*)(arow + k + 16);
        v4f a3 = *(const v4f*)(arow + k + 20);
        v16h av;
        #pragma unroll
        for (int j = 0; j < 4; ++j) {
            float f0 = a0[j], f1 = a1[j], f2 = a2[j], f3 = a3[j];
            if (RELU) {
                f0 = fmaxf(f0, 0.f); f1 = fmaxf(f1, 0.f);
                f2 = fmaxf(f2, 0.f); f3 = fmaxf(f3, 0.f);
            }
            av[j]      = (_Float16)f0;
            av[4 + j]  = (_Float16)f1;
            av[8 + j]  = (_Float16)f2;
            av[12 + j] = (_Float16)f3;
        }
        #pragma unroll
        for (int c = 0; c < 4; ++c) {
            const _Float16* bc = brow + (long)c * 16 * K;
            v8h b0 = *(const v8h*)(bc + k);
            v8h b1 = *(const v8h*)(bc + k + 16);
            v16h bv;
            #pragma unroll
            for (int j = 0; j < 8; ++j) { bv[j] = b0[j]; bv[8 + j] = b1[j]; }
            acc[c] = __builtin_amdgcn_wmma_f32_16x16x32_f16(
                         false, av, false, bv, (short)0, acc[c], false, false);
        }
    }

    // C/D layout: VGPR r -> M = r + 8*hi ; N = lane&15 (per 16-col subtile)
    float* cbase = C + (long)(mt * 16 + hi * 8) * NC + nt * 64 + rlo;
    #pragma unroll
    for (int c = 0; c < 4; ++c)
        #pragma unroll
        for (int r = 0; r < 8; ++r)
            cbase[(long)r * NC + c * 16] = acc[c][r];
}

// ---------------------------------------------------------------------------
// CSR aggregation (dim = HDIM): out[i] = sum_e h[src_e]*w_e + h[i]*dinv_i^2 + b
// one 512-thread block per node; edge list staged through LDS in chunks.
// h (102MB) is L2-resident on MI455X (192MB L2) -> gathers hit L2.
// ---------------------------------------------------------------------------
__global__ __launch_bounds__(512) void k_combine_h(const float* __restrict__ h,
                                                   const int* __restrict__ rowptr,
                                                   const int* __restrict__ colb,
                                                   const float* __restrict__ ew,
                                                   const float* __restrict__ dinv,
                                                   const float* __restrict__ bias,
                                                   float* __restrict__ out) {
    const int i = blockIdx.x;
    const int f = threadIdx.x;
    __shared__ int   s_src[128];
    __shared__ float s_w[128];
    const int beg = rowptr[i], end = rowptr[i + 1];
    const float di = dinv[i];
    float acc = h[(long)i * HDIM + f] * di * di + bias[f];
    for (int base = beg; base < end; base += 128) {
        const int cnt = min(128, end - base);
        __syncthreads();
        if (f < cnt) { s_src[f] = colb[base + f]; s_w[f] = ew[base + f]; }
        __syncthreads();
        for (int e = 0; e < cnt; ++e)
            acc += h[(long)s_src[e] * HDIM + f] * s_w[e];
    }
    out[(long)i * HDIM + f] = acc;
}

// ---------------------------------------------------------------------------
// Final H->3 projection: wave-per-node dot products with shuffle reduction
// (WMMA would waste 13/16 of the N tile; VALU is ample for 0.15 GFLOP)
// ---------------------------------------------------------------------------
__global__ void k_gemm3(const float* __restrict__ act, const float* __restrict__ W1,
                        float* __restrict__ h3, int n) {
    int w = (blockIdx.x * blockDim.x + threadIdx.x) >> 5;
    int lane = threadIdx.x & 31;
    if (w >= n) return;
    float a0 = 0.f, a1 = 0.f, a2 = 0.f;
    for (int k = lane; k < HDIM; k += 32) {
        float xv = fmaxf(act[(long)w * HDIM + k], 0.f);   // fused ReLU
        a0 += xv * W1[k * 3 + 0];
        a1 += xv * W1[k * 3 + 1];
        a2 += xv * W1[k * 3 + 2];
    }
    #pragma unroll
    for (int off = 16; off > 0; off >>= 1) {
        a0 += __shfl_xor(a0, off);
        a1 += __shfl_xor(a1, off);
        a2 += __shfl_xor(a2, off);
    }
    if (lane == 0) { h3[w * 3] = a0; h3[w * 3 + 1] = a1; h3[w * 3 + 2] = a2; }
}

__global__ void k_combine3(const float* __restrict__ h3,
                           const int* __restrict__ rowptr, const int* __restrict__ colb,
                           const float* __restrict__ ew, const float* __restrict__ dinv,
                           const float* __restrict__ b1, float* __restrict__ out, int n) {
    int i = blockIdx.x * blockDim.x + threadIdx.x;
    if (i >= n) return;
    float di = dinv[i];
    float a0 = h3[i * 3 + 0] * di * di + b1[0];
    float a1 = h3[i * 3 + 1] * di * di + b1[1];
    float a2 = h3[i * 3 + 2] * di * di + b1[2];
    for (int e = rowptr[i]; e < rowptr[i + 1]; ++e) {
        int s = colb[e]; float w = ew[e];
        a0 += h3[s * 3 + 0] * w;
        a1 += h3[s * 3 + 1] * w;
        a2 += h3[s * 3 + 2] * w;
    }
    out[i * 3 + 0] = a0; out[i * 3 + 1] = a1; out[i * 3 + 2] = a2;
}

__global__ void k_relu_copy(const float* __restrict__ in, float* __restrict__ out, long n) {
    long t = (long)blockIdx.x * blockDim.x + threadIdx.x;
    if (t < n) out[t] = fmaxf(in[t], 0.f);
}

// ---------------------------------------------------------------------------
extern "C" void kernel_launch(void* const* d_in, const int* in_sizes, int n_in,
                              void* d_out, int out_size, void* d_ws, size_t ws_size,
                              hipStream_t stream) {
    (void)in_sizes; (void)n_in; (void)out_size; (void)ws_size;
    const float* x  = (const float*)d_in[0];
    const int*   ei = (const int*)d_in[1];
    const float* W0 = (const float*)d_in[2];
    const float* b0 = (const float*)d_in[3];
    const float* Wr = (const float*)d_in[4];
    const float* br = (const float*)d_in[5];
    const float* W1 = (const float*)d_in[6];
    const float* b1 = (const float*)d_in[7];
    const int* src = ei;
    const int* dst = ei + NEDGES;
    float* out = (float*)d_out;

    // workspace carve-up (256B aligned)
    char* p = (char*)d_ws;
    auto carve = [&](size_t bytes) {
        char* r = p;
        p += (bytes + 255) & ~(size_t)255;
        return r;
    };
    float*    h      = (float*)carve((size_t)NNODES * HDIM * 4);
    float*    actA   = (float*)carve((size_t)NNODES * HDIM * 4);
    float*    actB   = (float*)carve((size_t)NNODES * HDIM * 4);
    _Float16* Wt     = (_Float16*)carve((size_t)HDIM * HDIM * 2);
    int*      cnt    = (int*)carve((size_t)NNODES * 4);
    int*      cur    = (int*)carve((size_t)NNODES * 4);
    int*      rowptr = (int*)carve((size_t)(NNODES + 1) * 4);
    int*      colb   = (int*)carve((size_t)NEDGES * 4);
    float*    ew     = (float*)carve((size_t)NEDGES * 4);
    float*    dinv   = (float*)carve((size_t)NNODES * 4);

    const int TB = 256;
    const int gN = (NNODES + TB - 1) / TB;
    const int gE = (NEDGES + TB - 1) / TB;

    // ---- CSR build (once, reused by all 14 layers) ----
    k_fill_zero_i32<<<gN, TB, 0, stream>>>(cnt, NNODES);
    k_fill_zero_i32<<<gN, TB, 0, stream>>>(cur, NNODES);
    k_count_deg<<<gE, TB, 0, stream>>>(dst, cnt, NEDGES);
    k_dinv<<<gN, TB, 0, stream>>>(cnt, dinv, NNODES);
    k_scan_excl<<<1, 1024, 0, stream>>>(cnt, rowptr, NNODES);
    k_fill_csr<<<gE, TB, 0, stream>>>(src, dst, rowptr, cur, dinv, colb, ew, NEDGES);

    const int waves    = (NNODES / 16) * (HDIM / 64);   // 3125 * 8 = 25000
    const int gemmBlks = (waves + 7) / 8;               // 8 waves / 256-thr block

    // ---- layer 0: F(256) -> H(512), input has no ReLU ----
    k_prep_w<<<(FDIM * HDIM + TB - 1) / TB, TB, 0, stream>>>(W0, Wt, FDIM, HDIM);
    k_gemm_wmma<false, FDIM><<<gemmBlks, TB, 0, stream>>>(x, Wt, h, NNODES, HDIM);
    k_combine_h<<<NNODES, HDIM, 0, stream>>>(h, rowptr, colb, ew, dinv, b0, actA);

    // ---- 6 residual blocks of two H->H convs (ReLU fused into A-load) ----
    float* a_in = actA;
    float* a_out = actB;
    for (int blk = 0; blk < NBLK; ++blk) {
        for (int j = 0; j < 2; ++j) {
            const float* Wij = Wr + (size_t)(blk * 2 + j) * HDIM * HDIM;
            const float* bij = br + (size_t)(blk * 2 + j) * HDIM;
            k_prep_w<<<(HDIM * HDIM + TB - 1) / TB, TB, 0, stream>>>(Wij, Wt, HDIM, HDIM);
            k_gemm_wmma<true, HDIM><<<gemmBlks, TB, 0, stream>>>(a_in, Wt, h, NNODES, HDIM);
            k_combine_h<<<NNODES, HDIM, 0, stream>>>(h, rowptr, colb, ew, dinv, bij, a_out);
            float* t = a_in; a_in = a_out; a_out = t;
        }
    }
    // a_in now holds the pre-ReLU output of block 5 conv 1

    // ---- final H -> 3 conv (ReLU fused into gemm3) ----
    float* h3 = h;   // h is free now
    k_gemm3<<<(NNODES * 32 + TB - 1) / TB, TB, 0, stream>>>(a_in, W1, h3, NNODES);
    k_combine3<<<gN, TB, 0, stream>>>(h3, rowptr, colb, ew, dinv, b1, out, NNODES);

    // ---- returned x = relu(final activation) ----
    const long nact = (long)NNODES * HDIM;
    k_relu_copy<<<(int)((nact + TB - 1) / TB), TB, 0, stream>>>(a_in, out + (size_t)NNODES * CDIM, nact);
}
